// mLSTMLayer_15814069584528
// MI455X (gfx1250) — matrix operand order repaired and checked by reading, not compiled, not run on hardware
//
#include <hip/hip_runtime.h>
#include <hip/hip_bf16.h>
#include <math.h>

typedef __attribute__((ext_vector_type(16))) _Float16 v16h;
typedef __attribute__((ext_vector_type(8)))  float    v8f;
typedef __attribute__((ext_vector_type(4)))  float    f4;
typedef __attribute__((ext_vector_type(4)))  _Float16 h4;
typedef __attribute__((ext_vector_type(8)))  _Float16 h8;

#define B_   2
#define S_   2048
#define E_   2048
#define H_   2688
#define NH_  4
#define DH_  672
#define KS_  4
#define NPH_ 672

__device__ __forceinline__ v8f wmma_f16(v16h a, v16h b, v8f c) {
  // D = A(16x32 f16) * B(32x16 f16) + C(16x16 f32)
  return __builtin_amdgcn_wmma_f32_16x16x32_f16(false, a, false, b, (short)0, c,
                                                false, false);
}

// A fragment: 16(M) x 32(K) f16, row-major source base[row*ld + k]
// Layout: lanes 0-15 = rows, element s holds K=s; lanes 16-31 K=16+s.
__device__ __forceinline__ v16h load_a_frag(const _Float16* base, int ld, int lane) {
  v16h a;
  int row = lane & 15;
  int hi  = (lane >> 4) * 16;
#pragma unroll
  for (int v = 0; v < 8; ++v) {
    int k0 = 2 * v + hi;
    a[2 * v]     = base[row * ld + k0];
    a[2 * v + 1] = base[row * ld + k0 + 1];
  }
  return a;
}

// B fragment: 32(K) x 16(N) f16, row-major source base[k*ld + col]
__device__ __forceinline__ v16h load_b_frag(const _Float16* base, int ld, int lane) {
  v16h b;
  int col  = lane & 15;
  int koff = (lane >> 4) * 16;
#pragma unroll
  for (int v = 0; v < 8; ++v) {
    b[2 * v]     = base[(koff + 2 * v) * ld + col];
    b[2 * v + 1] = base[(koff + 2 * v + 1) * ld + col];
  }
  return b;
}

// B fragment from transposed source: element(k,n) = base[n*ld + k]  (k^T for QK^T)
__device__ __forceinline__ v16h load_bT_frag(const _Float16* base, int ld, int lane) {
  v16h b;
  int col  = lane & 15;
  int koff = (lane >> 4) * 16;
#pragma unroll
  for (int v = 0; v < 8; ++v) {
    b[2 * v]     = base[col * ld + koff + 2 * v];
    b[2 * v + 1] = base[col * ld + koff + 2 * v + 1];
  }
  return b;
}

// ------------- generic fp32 GEMM via f16 WMMA (128x128 block, 8 waves) ---------
// Each wave: 64 rows x 32 cols = 8 WMMA tiles per K-panel from 6 fragment loads.
// Staging is split load-phase / store-phase so 8 b128 loads are in flight at once.
__global__ __launch_bounds__(256) void gemm_f32_wmma(
    const float* __restrict__ A, const float* __restrict__ Bm, float* __restrict__ C,
    int M, int N, int K) {
  __shared__ _Float16 As[128 * 40];    // 128 x 32 (+pad)
  __shared__ _Float16 Bs[32 * 136];    // 32 x 128 (+pad)
  const int tid  = threadIdx.x;
  const int lane = tid & 31, wid = tid >> 5;
  const int wm = wid >> 2, wn = wid & 3;
  const int m0 = blockIdx.y * 128, n0 = blockIdx.x * 128;
  v8f acc[4][2];
  v8f vzero = {};
#pragma unroll
  for (int i = 0; i < 4; ++i) { acc[i][0] = vzero; acc[i][1] = vzero; }
  for (int kk = 0; kk < K; kk += 32) {
    f4 va[4], vb[4];
#pragma unroll
    for (int i = 0; i < 4; ++i) {      // issue all A loads (b128)
      int idx = tid + i * 256;
      int r = idx >> 3, c4 = (idx & 7) * 4;
      va[i] = *(const f4*)&A[(size_t)(m0 + r) * K + kk + c4];
    }
#pragma unroll
    for (int i = 0; i < 4; ++i) {      // issue all B loads (b128)
      int idx = tid + i * 256;
      int r = idx >> 5, c4 = (idx & 31) * 4;
      vb[i] = *(const f4*)&Bm[(size_t)(kk + r) * N + n0 + c4];
    }
    if (kk + 32 < K) {                 // global_prefetch_b8 of next K-panel
      __builtin_prefetch(&A[(size_t)(m0 + (tid >> 1)) * K + kk + 32 + (tid & 1) * 16], 0, 3);
      __builtin_prefetch(&Bm[(size_t)(kk + 32 + (tid >> 3)) * N + n0 + (tid & 7) * 16], 0, 3);
    }
#pragma unroll
    for (int i = 0; i < 4; ++i) {      // convert + store A to LDS
      int idx = tid + i * 256;
      int r = idx >> 3, c4 = (idx & 7) * 4;
      *(h4*)&As[r * 40 + c4] = __builtin_convertvector(va[i], h4);
    }
#pragma unroll
    for (int i = 0; i < 4; ++i) {      // convert + store B to LDS
      int idx = tid + i * 256;
      int r = idx >> 5, c4 = (idx & 31) * 4;
      *(h4*)&Bs[r * 136 + c4] = __builtin_convertvector(vb[i], h4);
    }
    __syncthreads();
    v16h b0 = load_b_frag(Bs + wn * 32, 136, lane);
    v16h b1 = load_b_frag(Bs + wn * 32 + 16, 136, lane);
#pragma unroll
    for (int i = 0; i < 4; ++i) {
      v16h a = load_a_frag(As + (wm * 64 + i * 16) * 40, 40, lane);
      acc[i][0] = wmma_f16(a, b0, acc[i][0]);
      acc[i][1] = wmma_f16(a, b1, acc[i][1]);
    }
    __syncthreads();
  }
  int rl = (lane >> 4) * 8, cb = lane & 15;
#pragma unroll
  for (int i = 0; i < 4; ++i)
#pragma unroll
    for (int j = 0; j < 2; ++j)
#pragma unroll
      for (int v = 0; v < 8; ++v)
        C[(size_t)(m0 + wm * 64 + i * 16 + rl + v) * N + n0 + wn * 32 + j * 16 + cb] =
            acc[i][j][v];
}

// ------------- causal depthwise conv + SiLU + headwise q/k/v (4x4 blocks) ------
__global__ void conv_silu_qkv(const float* __restrict__ x_inner,
                              const float* __restrict__ conv_w, const float* __restrict__ conv_b,
                              const float* __restrict__ Wq, const float* __restrict__ Wk,
                              const float* __restrict__ Wv,
                              float* __restrict__ xca,
                              _Float16* __restrict__ qf, _Float16* __restrict__ kf,
                              _Float16* __restrict__ vf) {
  int gid = blockIdx.x * blockDim.x + threadIdx.x;
  int total = B_ * S_ * NPH_;
  if (gid >= total) return;
  int ph = gid % NPH_;
  int s  = (gid / NPH_) % S_;
  int b  = gid / (NPH_ * S_);
  size_t rowbase = (size_t)(b * S_ + s) * (2 * H_);
  float xm[4], xc[4];
#pragma unroll
  for (int d = 0; d < 4; ++d) {
    int ch = ph * 4 + d;
    xm[d] = x_inner[rowbase + ch];
    float acc = conv_b[ch];
#pragma unroll
    for (int w = 0; w < KS_; ++w) {
      int sp = s - (KS_ - 1) + w;
      if (sp >= 0) acc += x_inner[(size_t)(b * S_ + sp) * (2 * H_) + ch] * conv_w[w * H_ + ch];
    }
    float sg = 1.f / (1.f + __expf(-acc));
    xc[d] = acc * sg;                                  // silu
    xca[(size_t)(b * S_ + s) * H_ + ch] = xc[d];
  }
#pragma unroll
  for (int o = 0; o < 4; ++o) {
    float q = 0.f, k = 0.f, v = 0.f;
#pragma unroll
    for (int d = 0; d < 4; ++d) {
      q += xc[d] * Wq[ph * 16 + d * 4 + o];
      k += xc[d] * Wk[ph * 16 + d * 4 + o];
      v += xm[d] * Wv[ph * 16 + d * 4 + o];
    }
    size_t oi = (size_t)(b * S_ + s) * H_ + ph * 4 + o;
    qf[oi] = (_Float16)q; kf[oi] = (_Float16)k; vf[oi] = (_Float16)v;
  }
}

// ---------------- gate projections (N=4): block-per-row reduction --------------
__global__ __launch_bounds__(256) void gate_proj(
    const _Float16* __restrict__ qf, const _Float16* __restrict__ kf,
    const _Float16* __restrict__ vf,
    const float* __restrict__ W_ig, const float* __restrict__ b_ig,
    const float* __restrict__ W_fg, const float* __restrict__ b_fg,
    float* __restrict__ ipre, float* __restrict__ fpre) {
  __shared__ float red[256];
  int row = blockIdx.x;                 // b*S + s
  int b = row / S_, s = row % S_;
  int tid = threadIdx.x;
  float aI[NH_] = {0, 0, 0, 0}, aF[NH_] = {0, 0, 0, 0};
  size_t base = (size_t)row * H_;
  for (int idx = tid; idx < 3 * H_; idx += 256) {
    float val;
    if (idx < H_)          val = (float)qf[base + idx];
    else if (idx < 2 * H_) val = (float)kf[base + idx - H_];
    else                   val = (float)vf[base + idx - 2 * H_];
#pragma unroll
    for (int n = 0; n < NH_; ++n) {
      aI[n] += val * W_ig[(size_t)idx * NH_ + n];
      aF[n] += val * W_fg[(size_t)idx * NH_ + n];
    }
  }
  for (int n = 0; n < NH_; ++n) {
    red[tid] = aI[n]; __syncthreads();
    for (int st = 128; st > 0; st >>= 1) { if (tid < st) red[tid] += red[tid + st]; __syncthreads(); }
    if (tid == 0) ipre[(size_t)(b * NH_ + n) * S_ + s] = red[0] + b_ig[n];
    __syncthreads();
    red[tid] = aF[n]; __syncthreads();
    for (int st = 128; st > 0; st >>= 1) { if (tid < st) red[tid] += red[tid + st]; __syncthreads(); }
    if (tid == 0) fpre[(size_t)(b * NH_ + n) * S_ + s] = red[0] + b_fg[n];
    __syncthreads();
  }
}

// ------- sequential scan: cs = cumsum(logsigmoid(f)), m = cs + prefixmax(i-cs) --
__global__ void gate_scan(const float* __restrict__ ipre, const float* __restrict__ fpre,
                          float* __restrict__ cs, float* __restrict__ mst) {
  int bn = blockIdx.x;                  // b*NH + nh
  if (threadIdx.x != 0) return;
  float c = 0.f, a = -1e30f;
  size_t base = (size_t)bn * S_;
  for (int s = 0; s < S_; ++s) {
    float f = fpre[base + s];
    float ls = (f >= 0.f) ? -log1pf(__expf(-f)) : (f - log1pf(__expf(f)));
    c += ls;
    a = fmaxf(a, ipre[base + s] - c);
    cs[base + s]  = c;
    mst[base + s] = c + a;              // m_i = cs_i + max_{j<=i}(ipre_j - cs_j)
  }
}

// ---------------- fused mLSTM attention: 16 query rows per block ---------------
__global__ __launch_bounds__(192) void mlstm_attn(
    const _Float16* __restrict__ qf, const _Float16* __restrict__ kf,
    const _Float16* __restrict__ vf,
    const float* __restrict__ cs_g, const float* __restrict__ ip_g,
    const float* __restrict__ m_g, float* __restrict__ hout) {
  extern __shared__ char smem[];
  const int LDQ = 680, LDK = 680, LDV = 680, LDSC = 40;
  _Float16* qs  = (_Float16*)smem;                 // 16 x 680
  _Float16* ks  = qs + 16 * LDQ;                   // 32 x 680
  _Float16* vs  = ks + 32 * LDK;                   // 32 x 680
  float*    scf = (float*)(vs + 32 * LDV);         // 16 x 32 fp32 scores
  _Float16* sch = (_Float16*)(scf + 16 * 32);      // 16 x 40 f16 scores
  float* csi  = (float*)(sch + 16 * LDSC);
  float* mi   = csi + 16;
  float* ssum = mi + 16;
  float* rinv = ssum + 16;

  const int tid  = threadIdx.x;
  const int lane = tid & 31, wid = tid >> 5;
  const int it = blockIdx.x;
  const int bn = blockIdx.y;                       // b*NH + nh
  const int b = bn / NH_, nh = bn % NH_;
  const int i0 = it * 16;
  const float scale = rsqrtf((float)DH_);
  const size_t qkvhead = (size_t)b * S_ * H_ + (size_t)nh * DH_;
  const size_t csbase  = (size_t)bn * S_;

  {                                                // stage q tile (batched b128)
    h8 qreg[7]; int rr[7], cc[7];
#pragma unroll
    for (int i = 0; i < 7; ++i) {
      int idx = tid + i * 192;                     // 16*84 = 1344 = 7*192
      rr[i] = idx / 84; cc[i] = (idx % 84) * 8;
      qreg[i] = *(const h8*)&qf[qkvhead + (size_t)(i0 + rr[i]) * H_ + cc[i]];
    }
#pragma unroll
    for (int i = 0; i < 7; ++i)
      *(h8*)&qs[rr[i] * LDQ + cc[i]] = qreg[i];
  }
  if (tid < 16) {
    csi[tid]  = cs_g[csbase + i0 + tid];
    mi[tid]   = m_g[csbase + i0 + tid];
    ssum[tid] = 0.f;
  }
  __syncthreads();

  v8f hzero = {};
  v8f hacc[7];
#pragma unroll
  for (int t = 0; t < 7; ++t) hacc[t] = hzero;

  const int nchunks = ((it * 16 + 15) >> 5) + 1;     // causal: j chunks of 32
  for (int jc = 0; jc < nchunks; ++jc) {
    int j0 = jc * 32;
#pragma unroll
    for (int p = 0; p < 2; ++p) {                    // stage k/v: 2 batches of 7
      h8 kreg[7], vreg[7]; int rr[7], cc[7];
#pragma unroll
      for (int i = 0; i < 7; ++i) {                  // 32*84 = 2688 = 14*192
        int idx = tid + (p * 7 + i) * 192;
        rr[i] = idx / 84; cc[i] = (idx % 84) * 8;
        size_t src = qkvhead + (size_t)(j0 + rr[i]) * H_ + cc[i];
        kreg[i] = *(const h8*)&kf[src];
        vreg[i] = *(const h8*)&vf[src];
      }
#pragma unroll
      for (int i = 0; i < 7; ++i) {
        *(h8*)&ks[rr[i] * LDK + cc[i]] = kreg[i];
        *(h8*)&vs[rr[i] * LDV + cc[i]] = vreg[i];
      }
    }
    __syncthreads();
    if (wid == 0) {                                  // wave 0: scores + stabilize
      int hi2 = lane >> 4;
      int clb = lane & 15;
#pragma unroll
      for (int t = 0; t < 2; ++t) {
        v8f acc = {};
        for (int kk = 0; kk < DH_ / 32; ++kk) {
          v16h a  = load_a_frag(qs + kk * 32, LDQ, lane);
          v16h bb = load_bT_frag(ks + (t * 16) * LDK + kk * 32, LDK, lane);
          acc = wmma_f16(a, bb, acc);
        }
        int cl = t * 16 + clb;
        int cg = j0 + cl;
        float csj = cs_g[csbase + cg];
        float ipj = ip_g[csbase + cg];
#pragma unroll
        for (int v = 0; v < 8; ++v) {
          int rl = v + hi2 * 8;
          float val = 0.f;
          if (cg <= i0 + rl)
            val = acc[v] * scale * __expf(csi[rl] - csj + ipj - mi[rl]);
          scf[rl * 32 + cl]   = val;
          sch[rl * LDSC + cl] = (_Float16)val;
        }
      }
    }
    __syncthreads();
    if (tid < 16) {                                  // row sums for normalizer
      float s = 0.f;
      for (int c = 0; c < 32; ++c) s += scf[tid * 32 + c];
      ssum[tid] += s;
    }
    v16h a = load_a_frag(sch, LDSC, lane);           // C tile as A (16x32)
#pragma unroll
    for (int t = 0; t < 7; ++t) {                    // each wave: 7 of 42 DH tiles
      int dh0 = (wid * 7 + t) * 16;
      v16h bb = load_b_frag(vs + dh0, LDV, lane);
      hacc[t] = wmma_f16(a, bb, hacc[t]);
    }
    __syncthreads();
  }
  if (tid < 16)
    rinv[tid] = 1.f / (fmaxf(fabsf(ssum[tid]), __expf(-mi[tid])) + 1e-6f);
  __syncthreads();
  int hi3 = lane >> 4;
  int cl  = lane & 15;
#pragma unroll
  for (int t = 0; t < 7; ++t) {
    int dh0 = (wid * 7 + t) * 16;
#pragma unroll
    for (int v = 0; v < 8; ++v) {
      int rl = v + hi3 * 8;
      hout[qkvhead + (size_t)(i0 + rl) * H_ + dh0 + cl] = hacc[t][v] * rinv[rl];
    }
  }
}

// -------------- multi-head layernorm + skip + SiLU(z) gate (in place) ----------
__global__ __launch_bounds__(256) void mhnorm_skip_gate(
    const float* __restrict__ x_inner, const float* __restrict__ xca,
    const float* __restrict__ norm_w, const float* __restrict__ skip,
    float* __restrict__ h) {
  __shared__ float r1[256], r2[256];
  int blk = blockIdx.x;                  // (b*S+s)*NH + nh
  int nh  = blk % NH_;
  int row = blk / NH_;
  int tid = threadIdx.x;
  size_t hb = (size_t)row * H_ + nh * DH_;
  float s = 0.f, ss = 0.f;
  for (int i = tid; i < DH_; i += 256) {
    float v = h[hb + i]; s += v; ss += v * v;
  }
  r1[tid] = s; r2[tid] = ss; __syncthreads();
  for (int st = 128; st > 0; st >>= 1) {
    if (tid < st) { r1[tid] += r1[tid + st]; r2[tid] += r2[tid + st]; }
    __syncthreads();
  }
  float mu   = r1[0] / (float)DH_;
  float var  = r2[0] / (float)DH_ - mu * mu;
  float rstd = rsqrtf(var + 1e-5f);
  for (int i = tid; i < DH_; i += 256) {
    int ch = nh * DH_ + i;
    float hn  = (h[hb + i] - mu) * rstd * norm_w[ch];
    float hk  = hn + skip[ch] * xca[(size_t)row * H_ + ch];
    float z   = x_inner[(size_t)row * (2 * H_) + H_ + ch];
    float sil = z / (1.f + __expf(-z));
    h[hb + i] = hk * sil;
  }
}

extern "C" void kernel_launch(void* const* d_in, const int* in_sizes, int n_in,
                              void* d_out, int out_size, void* d_ws, size_t ws_size,
                              hipStream_t stream) {
  (void)in_sizes; (void)n_in; (void)out_size; (void)ws_size;
  const float* x      = (const float*)d_in[0];
  const float* W_in   = (const float*)d_in[1];
  const float* conv_w = (const float*)d_in[2];
  const float* conv_b = (const float*)d_in[3];
  const float* Wq     = (const float*)d_in[4];
  const float* Wk     = (const float*)d_in[5];
  const float* Wv     = (const float*)d_in[6];
  const float* W_ig   = (const float*)d_in[7];
  const float* b_ig   = (const float*)d_in[8];
  const float* W_fg   = (const float*)d_in[9];
  const float* b_fg   = (const float*)d_in[10];
  const float* norm_w = (const float*)d_in[11];
  const float* skip   = (const float*)d_in[12];
  const float* W_out  = (const float*)d_in[13];
  float* out = (float*)d_out;

  char* ws = (char*)d_ws;
  size_t off = 0;
  auto take = [&](size_t bytes) {
    char* p = ws + off;
    off = (off + bytes + 255) & ~(size_t)255;
    return p;
  };
  const size_t MS = (size_t)B_ * S_;                  // 4096 rows
  float*    x_inner = (float*)take(MS * 2 * H_ * 4);
  float*    xca     = (float*)take(MS * H_ * 4);
  _Float16* qf      = (_Float16*)take(MS * H_ * 2);
  _Float16* kf      = (_Float16*)take(MS * H_ * 2);
  _Float16* vf      = (_Float16*)take(MS * H_ * 2);
  float*    ipre    = (float*)take((size_t)B_ * NH_ * S_ * 4);
  float*    fpre    = (float*)take((size_t)B_ * NH_ * S_ * 4);
  float*    cs      = (float*)take((size_t)B_ * NH_ * S_ * 4);
  float*    mst     = (float*)take((size_t)B_ * NH_ * S_ * 4);
  float*    hbuf    = (float*)take(MS * H_ * 4);

  // 1. x @ W_in  (4096 x 2048 @ 2048 x 5376)
  gemm_f32_wmma<<<dim3((2 * H_) / 128, (unsigned)(MS / 128)), 256, 0, stream>>>(
      x, W_in, x_inner, (int)MS, 2 * H_, E_);
  // 2. causal conv + SiLU + headwise q/k/v
  {
    int total = B_ * S_ * NPH_;
    conv_silu_qkv<<<(total + 255) / 256, 256, 0, stream>>>(
        x_inner, conv_w, conv_b, Wq, Wk, Wv, xca, qf, kf, vf);
  }
  // 3. gate projections + scan
  gate_proj<<<(unsigned)MS, 256, 0, stream>>>(qf, kf, vf, W_ig, b_ig, W_fg, b_fg, ipre, fpre);
  gate_scan<<<B_ * NH_, 32, 0, stream>>>(ipre, fpre, cs, mst);
  // 4. fused mLSTM attention (dynamic LDS ~112 KB; WGP supports 320 KB)
  {
    size_t sm = (size_t)(16 * 680 + 2 * 32 * 680 + 16 * 40) * sizeof(_Float16) +
                (size_t)(16 * 32 + 64) * sizeof(float);
    mlstm_attn<<<dim3(S_ / 16, B_ * NH_), 192, sm, stream>>>(
        qf, kf, vf, cs, ipre, mst, hbuf);
  }
  // 5. multi-head layernorm + skip + SiLU(z) gate (in place on hbuf)
  mhnorm_skip_gate<<<(unsigned)(MS * NH_), 256, 0, stream>>>(x_inner, xca, norm_w, skip, hbuf);
  // 6. hs @ W_out  (4096 x 2688 @ 2688 x 2048)
  gemm_f32_wmma<<<dim3(E_ / 128, (unsigned)(MS / 128)), 256, 0, stream>>>(
      hbuf, W_out, out, (int)MS, E_, H_);
}
